// ModulatedLinear_39444979646803
// MI455X (gfx1250) — matrix operand order repaired and checked by reading, not compiled, not run on hardware
//
#include <hip/hip_runtime.h>

// ---------------------------------------------------------------------------
// ModulatedLinear on MI455X (gfx1250): split-bf16 WMMA GEMM, double-buffered.
//   Phase A: Wn = (1/32)*weight*y, demodulated; stored as bf16 hi+lo planes.
//   Phase B: Out_b = Wn_b x E_b via ah*bh + ah*bl + al*bh with
//            V_WMMA_F32_16X16X32_BF16 (f32 accum, ~1e-5 rel err vs fp32).
//   Pipeline: LDS double buffer; A tiles via GLOBAL_LOAD_ASYNC_TO_LDS_B128
//   (ASYNCcnt), B tiles via register prefetch + f32->bf16 conversion that
//   co-executes with the XDL WMMAs.
// ---------------------------------------------------------------------------

typedef __bf16 v16bf __attribute__((ext_vector_type(16)));
typedef float  v8f   __attribute__((ext_vector_type(8)));

#define IN_F  1024
#define OUT_F 1024
#define BATCH 16
#define LPOS  4096

#define BM  128
#define BN  128
#define BK  64
#define LDK 72                 // 64+8 pad: 144B row stride -> conflict-free
#define NT  (IN_F / BK)        // 16 K-tiles
#define ATILE (BM * LDK)       // elems per A plane per buffer
#define BTILE (BN * LDK)

#if __has_builtin(__builtin_amdgcn_global_load_async_to_lds_b128)
#define HAVE_ASYNC 1
// Builtin signature (from hipcc diagnostic): param0 = global ("__device__")
// pointer to  __attribute__((vector_size(16))) int,  param1 = LDS pointer.
typedef int vec4i __attribute__((vector_size(16)));
typedef __attribute__((address_space(1))) vec4i* as1_v4i;
typedef __attribute__((address_space(3))) vec4i* as3_v4i;
#else
#define HAVE_ASYNC 0
#endif

__device__ __forceinline__ void wait_async0() {
#if HAVE_ASYNC
#if __has_builtin(__builtin_amdgcn_s_wait_asynccnt)
  __builtin_amdgcn_s_wait_asynccnt(0);
#else
  asm volatile("s_wait_asynccnt 0x0" ::: "memory");
#endif
#endif
}

// ---- f32 <-> bf16 helpers (RNE, residual split) ---------------------------
__device__ __forceinline__ unsigned short f32_to_bf16(float f) {
  unsigned u = __float_as_uint(f);
  u += 0x7FFFu + ((u >> 16) & 1u);
  return (unsigned short)(u >> 16);
}
__device__ __forceinline__ float bf16_to_f32(unsigned short h) {
  return __uint_as_float(((unsigned)h) << 16);
}
__device__ __forceinline__ void split_bf16(float f, unsigned short& hi,
                                           unsigned short& lo) {
  hi = f32_to_bf16(f);
  lo = f32_to_bf16(f - bf16_to_f32(hi));
}

struct u4x2 { uint4 a, b; };
__device__ __forceinline__ v16bf frag_from(const unsigned short* p0,
                                           const unsigned short* p1) {
  u4x2 t;
  t.a = *(const uint4*)p0;
  t.b = *(const uint4*)p1;
  return __builtin_bit_cast(v16bf, t);
}
__device__ __forceinline__ v8f wmma_bf16(v16bf a, v16bf b, v8f c) {
  return __builtin_amdgcn_wmma_f32_16x16x32_bf16(
      false, a, false, b, (short)0, c, false, false);
}

// ------------------------- Phase A: weight modulation ----------------------
__global__ __launch_bounds__(256) void modulate_weights(
    const float* __restrict__ weight, const float* __restrict__ y,
    unsigned short* __restrict__ Whi, unsigned short* __restrict__ Wlo) {
  const int o   = blockIdx.x;
  const int b   = blockIdx.y;
  const int tid = threadIdx.x;
  const float scale = 0.03125f;          // 1/sqrt(1024)

  const float* wrow = weight + (size_t)o * IN_F;
  const float* yrow = y + (size_t)b * IN_F;

  float4 wv = *(const float4*)(wrow + (tid << 2));
  float4 yv = *(const float4*)(yrow + (tid << 2));
  float w[4] = { scale * wv.x * yv.x, scale * wv.y * yv.y,
                 scale * wv.z * yv.z, scale * wv.w * yv.w };
  float ss = w[0]*w[0] + w[1]*w[1] + w[2]*w[2] + w[3]*w[3];

  __shared__ float red[256];
  red[tid] = ss;
  __syncthreads();
  for (int s = 128; s > 0; s >>= 1) {
    if (tid < s) red[tid] += red[tid + s];
    __syncthreads();
  }
  float ssum = red[0] + 1e-8f;
  float r = rsqrtf(ssum);
  r = r * (1.5f - 0.5f * ssum * r * r);

  unsigned short h[4], l[4];
#pragma unroll
  for (int i = 0; i < 4; ++i) split_bf16(w[i] * r, h[i], l[i]);

  const size_t base = ((size_t)b * OUT_F + o) * IN_F + (tid << 2);
  uint2 ph, pl;
  ph.x = (unsigned)h[0] | ((unsigned)h[1] << 16);
  ph.y = (unsigned)h[2] | ((unsigned)h[3] << 16);
  pl.x = (unsigned)l[0] | ((unsigned)l[1] << 16);
  pl.y = (unsigned)l[2] | ((unsigned)l[3] << 16);
  *(uint2*)(Whi + base) = ph;
  *(uint2*)(Wlo + base) = pl;
}

// ------------------------- Phase B: pipelined split-bf16 GEMM --------------
__global__ __launch_bounds__(256) void mod_gemm_bf16x2(
    const unsigned short* __restrict__ Whi,
    const unsigned short* __restrict__ Wlo,
    const float* __restrict__ E,
    float* __restrict__ Out) {
  extern __shared__ ulonglong2 smem_raw[];            // 16B-aligned dynamic LDS
  unsigned short* smem = (unsigned short*)smem_raw;
  unsigned short* sAhi = smem;                        // [2][ATILE]
  unsigned short* sAlo = sAhi + 2 * ATILE;            // [2][ATILE]
  unsigned short* sBhi = sAlo + 2 * ATILE;            // [2][BTILE]  ([N][K])
  unsigned short* sBlo = sBhi + 2 * BTILE;            // [2][BTILE]

  const int tid  = threadIdx.x;
  const int lane = tid & 31;
  const int wave = tid >> 5;
  const int hh   = lane >> 4;
  const int l15  = lane & 15;
  const int wm   = wave >> 2;     // 0..1 (M)
  const int wn   = wave & 3;      // 0..3 (N)

  const int nBase = blockIdx.x * BN;
  const int mBase = blockIdx.y * BM;
  const int b     = blockIdx.z;

  const unsigned short* AhiB = Whi + (size_t)b * OUT_F * IN_F;
  const unsigned short* AloB = Wlo + (size_t)b * OUT_F * IN_F;
  const float*          Bb   = E   + (size_t)b * IN_F  * LPOS;
  float*                Cb   = Out + (size_t)b * OUT_F * LPOS;

  v8f acc[4][2];
#pragma unroll
  for (int i = 0; i < 4; ++i)
#pragma unroll
    for (int j = 0; j < 2; ++j) acc[i][j] = (v8f){0.f,0.f,0.f,0.f,0.f,0.f,0.f,0.f};

  // A staging: 4 chunks of 8 bf16 per thread (per plane)
  int aRow[4], aCol[4];
#pragma unroll
  for (int j = 0; j < 4; ++j) {
    int c = tid + 256 * j;
    aRow[j] = c >> 3;
    aCol[j] = (c & 7) << 3;
  }
  // B staging: one N-column per thread, 8 quads of 4 consecutive K
  const int colB = tid & 127;
  const int kg   = tid >> 7;

  float vB[32];
#if !HAVE_ASYNC
  uint4 pAh[4], pAl[4];
#endif

  auto issueA = [&](int buf, int k0) {
#pragma unroll
    for (int j = 0; j < 4; ++j) {
      size_t g = (size_t)(mBase + aRow[j]) * IN_F + k0 + aCol[j];
#if HAVE_ASYNC
      int off = buf * ATILE + aRow[j] * LDK + aCol[j];
      __builtin_amdgcn_global_load_async_to_lds_b128(
          (as1_v4i)(AhiB + g), (as3_v4i)(&sAhi[off]), 0, 0);
      __builtin_amdgcn_global_load_async_to_lds_b128(
          (as1_v4i)(AloB + g), (as3_v4i)(&sAlo[off]), 0, 0);
#else
      pAh[j] = *(const uint4*)(AhiB + g);
      pAl[j] = *(const uint4*)(AloB + g);
#endif
    }
  };
  auto commitA = [&](int buf) {
#if !HAVE_ASYNC
#pragma unroll
    for (int j = 0; j < 4; ++j) {
      int off = buf * ATILE + aRow[j] * LDK + aCol[j];
      *(uint4*)(&sAhi[off]) = pAh[j];
      *(uint4*)(&sAlo[off]) = pAl[j];
    }
#else
    (void)buf;
#endif
  };
  auto loadB = [&](int k0) {
#pragma unroll
    for (int j = 0; j < 8; ++j) {
      int kk = (kg + 2 * j) << 2;
      const float* src = Bb + (size_t)(k0 + kk) * LPOS + nBase + colB;
      vB[4*j+0] = src[0];
      vB[4*j+1] = src[(size_t)LPOS];
      vB[4*j+2] = src[(size_t)2 * LPOS];
      vB[4*j+3] = src[(size_t)3 * LPOS];
    }
  };
  auto commitB = [&](int buf) {
#pragma unroll
    for (int j = 0; j < 8; ++j) {
      int kk = (kg + 2 * j) << 2;
      unsigned short h0,l0,h1,l1,h2,l2,h3,l3;
      split_bf16(vB[4*j+0], h0, l0);
      split_bf16(vB[4*j+1], h1, l1);
      split_bf16(vB[4*j+2], h2, l2);
      split_bf16(vB[4*j+3], h3, l3);
      uint2 ph, pl;
      ph.x = (unsigned)h0 | ((unsigned)h1 << 16);
      ph.y = (unsigned)h2 | ((unsigned)h3 << 16);
      pl.x = (unsigned)l0 | ((unsigned)l1 << 16);
      pl.y = (unsigned)l2 | ((unsigned)l3 << 16);
      int off = buf * BTILE + colB * LDK + kk;
      *(uint2*)(&sBhi[off]) = ph;
      *(uint2*)(&sBlo[off]) = pl;
    }
  };
  auto compute = [&](int buf) {
#pragma unroll
    for (int ks = 0; ks < BK; ks += 32) {
      v16bf bh[2], bl[2];
#pragma unroll
      for (int tj = 0; tj < 2; ++tj) {
        int n    = wn * 32 + tj * 16 + l15;
        int base = buf * BTILE + n * LDK + ks + 8 * hh;
        bh[tj] = frag_from(&sBhi[base], &sBhi[base + 16]);
        bl[tj] = frag_from(&sBlo[base], &sBlo[base + 16]);
      }
#pragma unroll
      for (int ti = 0; ti < 4; ++ti) {
        int m    = wm * 64 + ti * 16 + l15;
        int base = buf * ATILE + m * LDK + ks + 8 * hh;
        v16bf ah = frag_from(&sAhi[base], &sAhi[base + 16]);
        v16bf al = frag_from(&sAlo[base], &sAlo[base + 16]);
        acc[ti][0] = wmma_bf16(ah, bh[0], acc[ti][0]);   // hi*hi
        acc[ti][1] = wmma_bf16(ah, bh[1], acc[ti][1]);
        acc[ti][0] = wmma_bf16(ah, bl[0], acc[ti][0]);   // hi*lo
        acc[ti][1] = wmma_bf16(ah, bl[1], acc[ti][1]);
        acc[ti][0] = wmma_bf16(al, bh[0], acc[ti][0]);   // lo*hi
        acc[ti][1] = wmma_bf16(al, bh[1], acc[ti][1]);
      }
    }
  };

  // ---------------- software pipeline, one barrier per K-tile --------------
  issueA(0, 0);
  loadB(0);
  commitB(0);
  commitA(0);
  wait_async0();
  __syncthreads();

  for (int t = 0; t < NT; ++t) {
    const int cur  = t & 1;
    const int nxt  = cur ^ 1;
    const bool more = (t + 1 < NT);
    if (more) {
      issueA(nxt, (t + 1) * BK);    // async -> LDS for next tile
      loadB((t + 1) * BK);          // global f32 loads in flight over compute
    }
    compute(cur);
    if (more) {
      commitB(nxt);                 // convert + ds_store (co-exec w/ tail)
      commitA(nxt);
      wait_async0();
      __syncthreads();
    }
  }

  // ---- epilogue: VGPR r holds row (tile_m + r + 8*half) -------------------
#pragma unroll
  for (int ti = 0; ti < 4; ++ti) {
    int rowBase = mBase + wm * 64 + ti * 16 + 8 * hh;
#pragma unroll
    for (int tj = 0; tj < 2; ++tj) {
      int col = nBase + wn * 32 + tj * 16 + l15;
      float* cp = Cb + (size_t)rowBase * LPOS + col;
#pragma unroll
      for (int r = 0; r < 8; ++r)
        cp[(size_t)r * LPOS] = acc[ti][tj][r];
    }
  }
}

// ---------------------------------------------------------------------------
extern "C" void kernel_launch(void* const* d_in, const int* in_sizes, int n_in,
                              void* d_out, int out_size, void* d_ws, size_t ws_size,
                              hipStream_t stream) {
  const float* Efou   = (const float*)d_in[0];  // [16, 1024, 4096]
  const float* y      = (const float*)d_in[1];  // [16, 1024]
  const float* weight = (const float*)d_in[2];  // [1, 1024, 1024, 1]
  float* out = (float*)d_out;

  unsigned short* Whi = (unsigned short*)d_ws;                 // 32 MB
  unsigned short* Wlo = Whi + (size_t)BATCH * OUT_F * IN_F;    // 32 MB

  const int shmem = (4 * ATILE + 4 * BTILE) * (int)sizeof(unsigned short); // 144 KB
  (void)hipFuncSetAttribute((const void*)mod_gemm_bf16x2,
                            hipFuncAttributeMaxDynamicSharedMemorySize, shmem);

  dim3 gridA(OUT_F, BATCH);
  modulate_weights<<<gridA, 256, 0, stream>>>(weight, y, Whi, Wlo);

  dim3 gridB(LPOS / BN, OUT_F / BM, BATCH);     // (32, 8, 16)
  mod_gemm_bf16x2<<<gridB, 256, shmem, stream>>>(Whi, Wlo, Efou, out);
}
